// MultiHeadAttention_15607911154383
// MI455X (gfx1250) — matrix operand order repaired and checked by reading, not compiled, not run on hardware
//
#include <hip/hip_runtime.h>

typedef __bf16 bf16;
typedef __attribute__((ext_vector_type(8)))  __bf16 v8bf;
typedef __attribute__((ext_vector_type(16))) __bf16 v16bf;
typedef __attribute__((ext_vector_type(8)))  float  v8f;

#define D_   1024
#define H_   16
#define HD_  64
#define S_   2048
#define B_   2
#define NT_  4096          // B_*S_ tokens
#define KCH_ (D_ / 32)     // 32 K-chunks per GEMM
#define SCALE_ 0.125f      // 1/sqrt(64), folded into Q at QKV epilogue

static __device__ inline v8f zero8() {
  v8f z = {0.f,0.f,0.f,0.f,0.f,0.f,0.f,0.f};
  return z;
}

static __device__ inline v16bf cat8(v8bf lo, v8bf hi) {
  return __builtin_shufflevector(lo, hi, 0,1,2,3,4,5,6,7,8,9,10,11,12,13,14,15);
}

static __device__ inline v8f wmma_bf16(v16bf a, v16bf b, v8f c) {
  // D = A(16x32 bf16) x B(32x16 bf16) + C(16x16 f32)
  return __builtin_amdgcn_wmma_f32_16x16x32_bf16(false, a, false, b, (short)0, c, false, false);
}

// A-matrix 16x32 bf16 fragment: lane half hf -> elems 0-15 = K 16hf..16hf+15
static __device__ inline v16bf loadA16(const bf16* row, int kb, int hf) {
  return cat8(*(const v8bf*)(row + kb + 16 * hf),
              *(const v8bf*)(row + kb + 16 * hf + 8));
}

// load one 32-K chunk of fragments: 2 A tiles (32 rows) + 4 B tiles (64 cols)
static __device__ __forceinline__ void gemm_load(const bf16* a0, const bf16* a1,
                                                 const bf16* const brow[4],
                                                 int kb, int hf,
                                                 v16bf (&af)[2], v16bf (&bfr)[4]) {
  af[0] = loadA16(a0, kb, hf);
  af[1] = loadA16(a1, kb, hf);
  #pragma unroll
  for (int a = 0; a < 4; ++a) bfr[a] = *(const v16bf*)(brow[a] + kb);
}

static __device__ __forceinline__ void gemm_mma(const v16bf (&af)[2], const v16bf (&bfr)[4],
                                                v8f (&acc)[2][4]) {
  #pragma unroll
  for (int a = 0; a < 4; ++a) {
    acc[0][a] = wmma_bf16(af[0], bfr[a], acc[0][a]);
    acc[1][a] = wmma_bf16(af[1], bfr[a], acc[1][a]);
  }
}

// ---------------- conversion / transpose ----------------

__global__ void cvt_f32_bf16(const float* __restrict__ in, bf16* __restrict__ out, int n) {
  int i = blockIdx.x * blockDim.x + threadIdx.x;
  int stride = gridDim.x * blockDim.x;
  for (; i < n; i += stride) out[i] = (bf16)in[i];
}

// W [K x N] f32 row-major  ->  Wt [N x K] bf16 row-major
__global__ void transpose_to_bf16(const float* __restrict__ W, bf16* __restrict__ Wt,
                                  int K, int N) {
  int i = blockIdx.x * blockDim.x + threadIdx.x;
  int stride = gridDim.x * blockDim.x;
  int total = K * N;
  for (; i < total; i += stride) {
    int k = i / N, j = i - k * N;
    Wt[(size_t)j * K + k] = (bf16)W[i];
  }
}

// ---------------- QKV projection GEMM ----------------
// wave tile: 32 rows x 64 cols; ping-pong double buffering (no register copies)
__global__ void __launch_bounds__(256)
qkv_gemm(const bf16* __restrict__ X, const bf16* __restrict__ Wt,
         const float* __restrict__ bias,
         bf16* __restrict__ Q, bf16* __restrict__ Kd, bf16* __restrict__ Vt) {
  const int lane = threadIdx.x & 31;
  const int wv   = threadIdx.x >> 5;
  const int wid  = blockIdx.x * 8 + wv;
  const int NTIL = (3 * D_) / 64;          // 48 column tiles of 64
  const int mt = wid / NTIL;               // 0..127 (32 rows each)
  const int nt = wid - mt * NTIL;
  const int hf = lane >> 4;
  const int ln = lane & 15;

  v8f acc[2][4];
  #pragma unroll
  for (int m = 0; m < 2; ++m)
    #pragma unroll
    for (int a = 0; a < 4; ++a) acc[m][a] = zero8();

  const bf16* a0 = X + (size_t)(mt * 32 + ln) * D_;
  const bf16* a1 = X + (size_t)(mt * 32 + 16 + ln) * D_;
  const bf16* brow[4];
  #pragma unroll
  for (int a = 0; a < 4; ++a)
    brow[a] = Wt + (size_t)(nt * 64 + a * 16 + ln) * D_ + 16 * hf;

  v16bf afA[2], bfA[4], afB[2], bfB[4];
  gemm_load(a0, a1, brow, 0,  hf, afA, bfA);
  gemm_load(a0, a1, brow, 32, hf, afB, bfB);

  for (int kc = 0; kc < KCH_ - 2; kc += 2) {
    gemm_mma(afA, bfA, acc);
    gemm_load(a0, a1, brow, (kc + 2) * 32, hf, afA, bfA);
    gemm_mma(afB, bfB, acc);
    gemm_load(a0, a1, brow, (kc + 3) * 32, hf, afB, bfB);
  }
  gemm_mma(afA, bfA, acc);
  gemm_mma(afB, bfB, acc);

  #pragma unroll
  for (int a = 0; a < 4; ++a) {
    const int j  = nt * 64 + a * 16 + ln;  // column in [0,3072)
    const float bj = bias[j];
    const int sec = j >> 10;               // 0=Q 1=K 2=V
    const int c   = j & (D_ - 1);
    const int hh  = c >> 6, hd = c & 63;
    #pragma unroll
    for (int m = 0; m < 2; ++m) {
      #pragma unroll
      for (int r = 0; r < 8; ++r) {
        const int t = mt * 32 + m * 16 + r + 8 * hf;  // token
        const int b = t >> 11, s = t & (S_ - 1);
        float v = acc[m][a][r] + bj;
        if (sec == 0)      Q [((size_t)(b * H_ + hh) * S_ + s) * HD_ + hd] = (bf16)(v * SCALE_);
        else if (sec == 1) Kd[((size_t)(b * H_ + hh) * S_ + s) * HD_ + hd] = (bf16)v;
        else               Vt[((size_t)(b * H_ + hh) * HD_ + hd) * S_ + s] = (bf16)v;
      }
    }
  }
}

// ---------------- flash attention ----------------
// one wave = one (b,h) x 16-query block; 32-key tiles, ping-pong K prefetch,
// V loads issued before softmax so VALU work hides memory latency
__global__ void __launch_bounds__(256)
flash_attn(const bf16* __restrict__ Q, const bf16* __restrict__ Kd,
           const bf16* __restrict__ Vt, bf16* __restrict__ O) {
  __shared__ bf16 pLds[8][16 * 32];
  const int lane = threadIdx.x & 31;
  const int wv   = threadIdx.x >> 5;
  const int wid  = blockIdx.x * 8 + wv;
  const int hf = lane >> 4;
  const int ln = lane & 15;
  const int qt = wid & 127;                // query tile (16 rows)
  const int bh = wid >> 7;                 // b*H + h, 0..31
  const int qbase = qt * 16;

  const bf16* qb = Q  + ((size_t)bh * S_ + qbase) * HD_;
  const bf16* kb = Kd + (size_t)bh * S_ * HD_;
  const bf16* vb = Vt + (size_t)bh * HD_ * S_;

  // Q fragments (pre-scaled by 1/sqrt(HD) in qkv_gemm): two 16x32 A-matrices
  v16bf qf[2];
  #pragma unroll
  for (int c = 0; c < 2; ++c)
    qf[c] = loadA16(qb + (size_t)ln * HD_, c * 32, hf);

  v8f o[4];
  #pragma unroll
  for (int a = 0; a < 4; ++a) o[a] = zero8();
  float mrow[8], lrow[8];
  #pragma unroll
  for (int r = 0; r < 8; ++r) { mrow[r] = -__builtin_inff(); lrow[r] = 0.f; }

  // K^T B-fragments, ping-pong: kfA = even tiles, kfB = odd tiles
  v16bf kfA[2][2], kfB[2][2];
  #pragma unroll
  for (int t = 0; t < 2; ++t)
    #pragma unroll
    for (int c = 0; c < 2; ++c) {
      kfA[t][c] = *(const v16bf*)(kb + (size_t)(t * 16 + ln) * HD_ + c * 32 + 16 * hf);
      kfB[t][c] = *(const v16bf*)(kb + (size_t)(32 + t * 16 + ln) * HD_ + c * 32 + 16 * hf);
    }

  auto attn_step = [&](int kt, v16bf (&cur)[2][2]) {
    // V fragments for this tile: issue loads early, consumed after softmax
    v16bf vf[4];
    #pragma unroll
    for (int a = 0; a < 4; ++a)
      vf[a] = *(const v16bf*)(vb + (size_t)(a * 16 + ln) * S_ + kt * 32 + 16 * hf);

    // scores from already-resident K fragments
    v8f s0 = zero8(), s1 = zero8();
    #pragma unroll
    for (int c = 0; c < 2; ++c) {
      s0 = wmma_bf16(qf[c], cur[0][c], s0);
      s1 = wmma_bf16(qf[c], cur[1][c], s1);
    }

    // refill this buffer with tile kt+2 (wrap -> branch-free, always valid)
    const int ktn = (kt + 2) & (S_ / 32 - 1);
    #pragma unroll
    for (int t = 0; t < 2; ++t)
      #pragma unroll
      for (int c = 0; c < 2; ++c)
        cur[t][c] = *(const v16bf*)(kb + (size_t)(ktn * 32 + t * 16 + ln) * HD_ + c * 32 + 16 * hf);

    // online softmax; row r+8*hf lives across the 16 lanes of this half
    #pragma unroll
    for (int r = 0; r < 8; ++r) {
      float x0 = s0[r];
      float x1 = s1[r];
      float mx = fmaxf(x0, x1);
      #pragma unroll
      for (int msk = 8; msk >= 1; msk >>= 1) mx = fmaxf(mx, __shfl_xor(mx, msk, 16));
      float mnew  = fmaxf(mrow[r], mx);
      float alpha = __expf(mrow[r] - mnew);
      float p0 = __expf(x0 - mnew);
      float p1 = __expf(x1 - mnew);
      float ps = p0 + p1;
      #pragma unroll
      for (int msk = 8; msk >= 1; msk >>= 1) ps += __shfl_xor(ps, msk, 16);
      lrow[r] = lrow[r] * alpha + ps;
      mrow[r] = mnew;
      #pragma unroll
      for (int a = 0; a < 4; ++a) o[a][r] = o[a][r] * alpha;
      const int m = r + 8 * hf;
      pLds[wv][m * 32 + ln]      = (bf16)p0;
      pLds[wv][m * 32 + 16 + ln] = (bf16)p1;
    }

    // P (16x32) as A-matrix from LDS; O += P * V_tile(32x64)
    v16bf pf = loadA16(&pLds[wv][ln * 32], 0, hf);
    #pragma unroll
    for (int a = 0; a < 4; ++a)
      o[a] = wmma_bf16(pf, vf[a], o[a]);
  };

  for (int kt = 0; kt < S_ / 32; kt += 2) {
    attn_step(kt,     kfA);
    attn_step(kt + 1, kfB);
  }

  const int b = bh >> 4, hh = bh & 15;
  #pragma unroll
  for (int a = 0; a < 4; ++a) {
    #pragma unroll
    for (int r = 0; r < 8; ++r) {
      const int s = qbase + r + 8 * hf;
      float val = o[a][r] / lrow[r];
      O[((size_t)(b * S_ + s)) * D_ + hh * HD_ + a * 16 + ln] = (bf16)val;
    }
  }
}

// ---------------- output projection GEMM ----------------
__global__ void __launch_bounds__(256)
out_gemm(const bf16* __restrict__ A, const bf16* __restrict__ Wt,
         const float* __restrict__ bias, float* __restrict__ out) {
  const int lane = threadIdx.x & 31;
  const int wv   = threadIdx.x >> 5;
  const int wid  = blockIdx.x * 8 + wv;
  const int NTIL = D_ / 64;                // 16
  const int mt = wid / NTIL;               // 0..127 (32 rows)
  const int nt = wid - mt * NTIL;
  const int hf = lane >> 4;
  const int ln = lane & 15;

  v8f acc[2][4];
  #pragma unroll
  for (int m = 0; m < 2; ++m)
    #pragma unroll
    for (int a = 0; a < 4; ++a) acc[m][a] = zero8();

  const bf16* a0 = A + (size_t)(mt * 32 + ln) * D_;
  const bf16* a1 = A + (size_t)(mt * 32 + 16 + ln) * D_;
  const bf16* brow[4];
  #pragma unroll
  for (int a = 0; a < 4; ++a)
    brow[a] = Wt + (size_t)(nt * 64 + a * 16 + ln) * D_ + 16 * hf;

  v16bf afA[2], bfA[4], afB[2], bfB[4];
  gemm_load(a0, a1, brow, 0,  hf, afA, bfA);
  gemm_load(a0, a1, brow, 32, hf, afB, bfB);

  for (int kc = 0; kc < KCH_ - 2; kc += 2) {
    gemm_mma(afA, bfA, acc);
    gemm_load(a0, a1, brow, (kc + 2) * 32, hf, afA, bfA);
    gemm_mma(afB, bfB, acc);
    gemm_load(a0, a1, brow, (kc + 3) * 32, hf, afB, bfB);
  }
  gemm_mma(afA, bfA, acc);
  gemm_mma(afB, bfB, acc);

  #pragma unroll
  for (int a = 0; a < 4; ++a) {
    const int j = nt * 64 + a * 16 + ln;
    const float bj = bias[j];
    #pragma unroll
    for (int m = 0; m < 2; ++m) {
      #pragma unroll
      for (int r = 0; r < 8; ++r) {
        const int t = mt * 32 + m * 16 + r + 8 * hf;
        out[(size_t)t * D_ + j] = acc[m][a][r] + bj;
      }
    }
  }
}

// ---------------- launch ----------------

extern "C" void kernel_launch(void* const* d_in, const int* in_sizes, int n_in,
                              void* d_out, int out_size, void* d_ws, size_t ws_size,
                              hipStream_t stream) {
  (void)in_sizes; (void)n_in; (void)out_size; (void)ws_size;
  const float* x     = (const float*)d_in[0];
  const float* W_qkv = (const float*)d_in[1];
  const float* b_qkv = (const float*)d_in[2];
  const float* W_out = (const float*)d_in[3];
  const float* b_out = (const float*)d_in[4];
  float* out = (float*)d_out;

  char* ws = (char*)d_ws;
  bf16* x_bf   = (bf16*)(ws);                          //  8 MiB: x bf16 [4096,1024]
  bf16* wqkv_t = (bf16*)(ws + ((size_t)8  << 20));     //  6 MiB: Wqkv^T bf16 [3072,1024]
  bf16* wout_t = (bf16*)(ws + ((size_t)14 << 20));     //  2 MiB: Wout^T bf16 [1024,1024]
  bf16* q_ws   = (bf16*)(ws + ((size_t)16 << 20));     //  8 MiB: Q [B,H,S,HD] (pre-scaled)
  bf16* k_ws   = (bf16*)(ws + ((size_t)24 << 20));     //  8 MiB: K [B,H,S,HD]
  bf16* vt_ws  = (bf16*)(ws + ((size_t)32 << 20));     //  8 MiB: V^T [B,H,HD,S]
  bf16* at_ws  = (bf16*)(ws + ((size_t)40 << 20));     //  8 MiB: attn out bf16 [4096,1024]

  cvt_f32_bf16<<<1024, 256, 0, stream>>>(x, x_bf, NT_ * D_);
  transpose_to_bf16<<<1024, 256, 0, stream>>>(W_qkv, wqkv_t, D_, 3 * D_);
  transpose_to_bf16<<<1024, 256, 0, stream>>>(W_out, wout_t, D_, D_);

  // 4096/32 row tiles * 48 col tiles = 6144 waves / 8 per block
  qkv_gemm<<<768, 256, 0, stream>>>(x_bf, wqkv_t, b_qkv, q_ws, k_ws, vt_ws);
  // B*H*S/16 = 4096 waves / 8 per block
  flash_attn<<<512, 256, 0, stream>>>(q_ws, k_ws, vt_ws, at_ws);
  // 4096/32 row tiles * 16 col tiles = 2048 waves / 8 per block
  out_gemm<<<256, 256, 0, stream>>>(at_ws, wout_t, b_out, out);
}
